// GWN_1864015806547
// MI455X (gfx1250) — matrix operand order repaired and checked by reading, not compiled, not run on hardware
//
#include <hip/hip_runtime.h>

#define NN   50000
#define EE   800000
#define RR   3
#define FIN  256
#define CH   128
#define BB   4096

typedef __attribute__((ext_vector_type(16))) __bf16 v16bf;
typedef __attribute__((ext_vector_type(8)))  __bf16 v8bf;
typedef __attribute__((ext_vector_type(8)))  float  v8f;

// ---------------- elementwise conversion / utility kernels ----------------

__global__ __launch_bounds__(256) void cvt_bf16_kernel(const float* __restrict__ in,
                                                       __bf16* __restrict__ out, int n) {
    int i = blockIdx.x * blockDim.x + threadIdx.x;
    if (i < n) out[i] = (__bf16)in[i];
}

__global__ __launch_bounds__(256) void zero_i32_kernel(int* __restrict__ p, int n) {
    int i = blockIdx.x * blockDim.x + threadIdx.x;
    if (i < n) p[i] = 0;
}

__global__ __launch_bounds__(256) void deg_count_kernel(const int* __restrict__ col,
                                                        int* __restrict__ deg, int e) {
    int i = blockIdx.x * blockDim.x + threadIdx.x;
    if (i < e) atomicAdd(&deg[col[i]], 1);
}

__global__ __launch_bounds__(256) void dinv_kernel(const int* __restrict__ deg,
                                                   float* __restrict__ dinv, int n) {
    int i = blockIdx.x * blockDim.x + threadIdx.x;
    if (i < n) {
        int d = deg[i];
        dinv[i] = (d > 0) ? rsqrtf((float)d) : 0.0f;
    }
}

// Single-workgroup exclusive scan of integer degrees -> CSR row pointers.
__global__ __launch_bounds__(1024) void scan_kernel(const int* __restrict__ deg,
                                                    int* __restrict__ ptr, int n) {
    __shared__ int sdata[1024];
    __shared__ int srun;
    if (threadIdx.x == 0) srun = 0;
    __syncthreads();
    for (int base = 0; base < n; base += 1024) {
        int i = base + (int)threadIdx.x;
        int v = (i < n) ? deg[i] : 0;
        sdata[threadIdx.x] = v;
        __syncthreads();
        for (int offs = 1; offs < 1024; offs <<= 1) {
            int t = 0;
            if (threadIdx.x >= (unsigned)offs) t = sdata[threadIdx.x - offs];
            __syncthreads();
            if (threadIdx.x >= (unsigned)offs) sdata[threadIdx.x] += t;
            __syncthreads();
        }
        int run = srun;
        if (i < n) ptr[i] = run + sdata[threadIdx.x] - v;  // exclusive
        __syncthreads();
        if (threadIdx.x == 0) srun = run + sdata[1023];
        __syncthreads();
    }
    if (threadIdx.x == 0) ptr[n] = srun;
}

__global__ __launch_bounds__(256) void csr_build_kernel(const int* __restrict__ row,
                                                        const int* __restrict__ col,
                                                        const float* __restrict__ dinv,
                                                        const int* __restrict__ ptr,
                                                        int* __restrict__ cursor,
                                                        int* __restrict__ crow,
                                                        float* __restrict__ cw, int e) {
    int i = blockIdx.x * blockDim.x + threadIdx.x;
    if (i >= e) return;
    int c = col[i];
    int r = row[i];
    int pos = ptr[c] + atomicAdd(&cursor[c], 1);
    crow[pos] = r;
    cw[pos] = dinv[r] * dinv[c];
}

// Deterministic summation order: sort each adjacency segment by source row.
__global__ __launch_bounds__(256) void sort_csr_kernel(const int* __restrict__ ptr,
                                                       int* __restrict__ crow,
                                                       float* __restrict__ cw) {
    int i = blockIdx.x * blockDim.x + threadIdx.x;
    if (i >= NN) return;
    int s = ptr[i], e = ptr[i + 1];
    for (int j = s + 1; j < e; ++j) {
        int rk = crow[j];
        float wk = cw[j];
        int k = j - 1;
        while (k >= s && crow[k] > rk) {
            crow[k + 1] = crow[k];
            cw[k + 1] = cw[k];
            --k;
        }
        crow[k + 1] = rk;
        cw[k + 1] = wk;
    }
}

// ---------------- dual-output WMMA GEMM: phi{0,1} = X @ W{0,1}^T + b{0,1} ----------------
// One wave computes one 16x16 output tile for BOTH linears (A-tile reuse).
// 8 waves per block cover the full C=128 band of a 16-row stripe.

__global__ __launch_bounds__(256) void gemm_dual_kernel(
    const __bf16* __restrict__ xa,    // [NN, FIN] bf16
    const __bf16* __restrict__ wb0,   // [CH, FIN] bf16 (relation slice)
    const __bf16* __restrict__ wb1,   // [CH, FIN] bf16
    const float* __restrict__ bias0,  // [CH]
    const float* __restrict__ bias1,  // [CH]
    float* __restrict__ phi0,         // [NN, CH]
    float* __restrict__ phi1)         // [NN, CH]
{
    const int rowBase = blockIdx.x * 16;
    const int colBase = (threadIdx.x >> 5) * 16;
    const int lane = threadIdx.x & 31;
    const int mn = lane & 15;   // M for A, N for B/C/D
    const int kh = lane >> 4;   // lane-half selector

    const __bf16* aRow = xa + (size_t)(rowBase + mn) * FIN;
    const __bf16* b0Row = wb0 + (size_t)(colBase + mn) * FIN;
    const __bf16* b1Row = wb1 + (size_t)(colBase + mn) * FIN;

    const float bs0 = bias0[colBase + mn];
    const float bs1 = bias1[colBase + mn];
    v8f acc0, acc1;
#pragma unroll
    for (int i = 0; i < 8; ++i) { acc0[i] = bs0; acc1[i] = bs1; }

#pragma unroll
    for (int kb = 0; kb < FIN / 32; ++kb) {
        const int k0 = kb * 32;
        // A 16x32 bf16: lanes 0-15 hold K=0..15, lanes 16-31 hold K=16..31 (contiguous,
        // matching B) -> two contiguous 16B runs.
        v8bf aLo = *reinterpret_cast<const v8bf*>(aRow + k0 + 16 * kh);
        v8bf aHi = *reinterpret_cast<const v8bf*>(aRow + k0 + 16 * kh + 8);
        // B 32x16 bf16: lanes 0-15 hold K=0..15, lanes 16-31 hold K=16..31 (contiguous).
        v8bf b0Lo = *reinterpret_cast<const v8bf*>(b0Row + k0 + 16 * kh);
        v8bf b0Hi = *reinterpret_cast<const v8bf*>(b0Row + k0 + 16 * kh + 8);
        v8bf b1Lo = *reinterpret_cast<const v8bf*>(b1Row + k0 + 16 * kh);
        v8bf b1Hi = *reinterpret_cast<const v8bf*>(b1Row + k0 + 16 * kh + 8);
        v16bf a, bb0, bb1;
#pragma unroll
        for (int j = 0; j < 8; ++j) {
            a[j] = aLo[j];    a[8 + j] = aHi[j];
            bb0[j] = b0Lo[j]; bb0[8 + j] = b0Hi[j];
            bb1[j] = b1Lo[j]; bb1[8 + j] = b1Hi[j];
        }
        acc0 = __builtin_amdgcn_wmma_f32_16x16x32_bf16(false, a, false, bb0,
                                                       (short)0, acc0, false, false);
        acc1 = __builtin_amdgcn_wmma_f32_16x16x32_bf16(false, a, false, bb1,
                                                       (short)0, acc1, false, false);
    }

    const int n = colBase + mn;
#pragma unroll
    for (int i = 0; i < 8; ++i) {
        const int m = rowBase + i + 8 * kh;  // C/D layout: lanes 16-31 -> M += 8
        phi0[(size_t)m * CH + n] = acc0[i];
        phi1[(size_t)m * CH + n] = acc1[i];
    }
}

// ---------------- fused axpy + CSR SpMM propagate ----------------
// out[i,:] = alpha*a[i,:] + beta*b[i,:] + gamma * sum_e w[e] * src[crow[e],:]
// One wave per node; each lane owns 4 channels (float4).

__global__ __launch_bounds__(256) void prop_kernel(
    const float* __restrict__ a, const float* __restrict__ b,
    const float* __restrict__ src, float* __restrict__ out,
    const int* __restrict__ ptr, const int* __restrict__ rows,
    const float* __restrict__ wv, float alpha, float beta, float gamma)
{
    const int node = (int)((blockIdx.x * blockDim.x + threadIdx.x) >> 5);
    const int lane = threadIdx.x & 31;
    if (node >= NN) return;

    const size_t base = (size_t)node * CH + lane * 4;
    const float4 av = *reinterpret_cast<const float4*>(a + base);
    const float4 bv = *reinterpret_cast<const float4*>(b + base);
    float4 acc;
    acc.x = alpha * av.x + beta * bv.x;
    acc.y = alpha * av.y + beta * bv.y;
    acc.z = alpha * av.z + beta * bv.z;
    acc.w = alpha * av.w + beta * bv.w;

    const float4* src4 = reinterpret_cast<const float4*>(src);
    const int e0 = ptr[node], e1 = ptr[node + 1];
    for (int e = e0; e < e1; ++e) {
        if (e + 1 < e1)
            __builtin_prefetch(src4 + (size_t)rows[e + 1] * (CH / 4) + lane, 0, 0);
        const float w = gamma * wv[e];
        const float4 xv = src4[(size_t)rows[e] * (CH / 4) + lane];
        acc.x += w * xv.x;
        acc.y += w * xv.y;
        acc.z += w * xv.z;
        acc.w += w * xv.w;
    }
    *reinterpret_cast<float4*>(out + base) = acc;
}

__global__ __launch_bounds__(256) void gather_out_kernel(const float* __restrict__ xf,
                                                         const int* __restrict__ batch,
                                                         float* __restrict__ out, int r) {
    int idx = blockIdx.x * blockDim.x + threadIdx.x;
    if (idx >= BB * CH) return;
    int b = idx / CH;
    int c = idx - b * CH;
    out[(size_t)b * (RR * CH) + r * CH + c] = xf[(size_t)batch[b] * CH + c];
}

// ---------------- orchestration ----------------

extern "C" void kernel_launch(void* const* d_in, const int* in_sizes, int n_in,
                              void* d_out, int out_size, void* d_ws, size_t ws_size,
                              hipStream_t stream) {
    (void)in_sizes; (void)n_in; (void)out_size; (void)ws_size;
    const float* features = (const float*)d_in[0];
    const int*   edges    = (const int*)d_in[1];   // [R, 2, E]
    const int*   batch    = (const int*)d_in[2];   // [B]
    const float* l0w = (const float*)d_in[3];      // [R, C, F_IN]
    const float* l0b = (const float*)d_in[4];      // [R, C]
    const float* l1w = (const float*)d_in[5];
    const float* l1b = (const float*)d_in[6];
    float* out = (float*)d_out;

    char* base = (char*)d_ws;
    size_t off = 0;
    auto take = [&](size_t bytes) -> char* {
        off = (off + 255) & ~(size_t)255;
        char* p = base + off;
        off += bytes;
        return p;
    };
    __bf16* featbf = (__bf16*)take((size_t)NN * FIN * 2);
    __bf16* w0bf   = (__bf16*)take((size_t)RR * CH * FIN * 2);
    __bf16* w1bf   = (__bf16*)take((size_t)RR * CH * FIN * 2);
    int*    degi   = (int*)take((size_t)NN * 4);
    float*  dinv   = (float*)take((size_t)NN * 4);
    int*    ptr    = (int*)take((size_t)(NN + 1) * 4);
    int*    cursor = (int*)take((size_t)NN * 4);
    int*    crow   = (int*)take((size_t)EE * 4);
    float*  cw     = (float*)take((size_t)EE * 4);
    float*  S0     = (float*)take((size_t)NN * CH * 4);
    float*  S1     = (float*)take((size_t)NN * CH * 4);

    const int T = 256;
    const float dt = 0.25f;

    cvt_bf16_kernel<<<(NN * FIN + T - 1) / T, T, 0, stream>>>(features, featbf, NN * FIN);
    cvt_bf16_kernel<<<(RR * CH * FIN + T - 1) / T, T, 0, stream>>>(l0w, w0bf, RR * CH * FIN);
    cvt_bf16_kernel<<<(RR * CH * FIN + T - 1) / T, T, 0, stream>>>(l1w, w1bf, RR * CH * FIN);

    for (int r = 0; r < RR; ++r) {
        const int* rowe = edges + (size_t)r * 2 * EE;  // src
        const int* cole = rowe + EE;                   // dst

        zero_i32_kernel<<<(NN + T - 1) / T, T, 0, stream>>>(degi, NN);
        zero_i32_kernel<<<(NN + T - 1) / T, T, 0, stream>>>(cursor, NN);
        deg_count_kernel<<<(EE + T - 1) / T, T, 0, stream>>>(cole, degi, EE);
        dinv_kernel<<<(NN + T - 1) / T, T, 0, stream>>>(degi, dinv, NN);
        scan_kernel<<<1, 1024, 0, stream>>>(degi, ptr, NN);
        csr_build_kernel<<<(EE + T - 1) / T, T, 0, stream>>>(rowe, cole, dinv, ptr,
                                                             cursor, crow, cw, EE);
        sort_csr_kernel<<<(NN + T - 1) / T, T, 0, stream>>>(ptr, crow, cw);

        // phi0 -> S0, phi1 -> S1
        gemm_dual_kernel<<<NN / 16, 256, 0, stream>>>(
            featbf, w0bf + (size_t)r * CH * FIN, w1bf + (size_t)r * CH * FIN,
            l0b + (size_t)r * CH, l1b + (size_t)r * CH, S0, S1);

        const int pb = (NN * 32 + 255) / 256;
        // init: x1 = dt*phi1 + phi0 + (dt^2/2)*SpMM(phi0)   (self-loop weight 1 folded)
        prop_kernel<<<pb, 256, 0, stream>>>(S1, S0, S0, S1, ptr, crow, cw,
                                            dt, 1.0f, dt * dt * 0.5f);
        // steps: xn = 2*xc - xp + dt^2*SpMM(xc)   (self-loop weight 2 folded), xn -> xp buf
        prop_kernel<<<pb, 256, 0, stream>>>(S1, S0, S1, S0, ptr, crow, cw,
                                            2.0f, -1.0f, dt * dt);
        prop_kernel<<<pb, 256, 0, stream>>>(S0, S1, S0, S1, ptr, crow, cw,
                                            2.0f, -1.0f, dt * dt);
        prop_kernel<<<pb, 256, 0, stream>>>(S1, S0, S1, S0, ptr, crow, cw,
                                            2.0f, -1.0f, dt * dt);
        // final state in S0
        gather_out_kernel<<<(BB * CH + T - 1) / T, T, 0, stream>>>(S0, batch, out, r);
    }
}
